// BingramLanguageModel_19645180411976
// MI455X (gfx1250) — compile-verified, matrix operand order checked by reading, not run on hardware
//
#include <hip/hip_runtime.h>
#include <hip/hip_bf16.h>

// ---------------------------------------------------------------------------
// Model dims (compile-time constants from the reference)
// ---------------------------------------------------------------------------
#define Cdim 512
#define Tdim 2048
#define Bdim 4
#define Vdim 32000
#define Fdim 2048            // 4*C
#define Mrows (Bdim * Tdim)  // 8192

typedef __attribute__((ext_vector_type(16))) __bf16 v16bf;
typedef __attribute__((ext_vector_type(8)))  float  v8f;
typedef __attribute__((ext_vector_type(4)))  unsigned int u32x4;
typedef __attribute__((ext_vector_type(8)))  int  i32x8;
typedef __attribute__((ext_vector_type(4)))  int  i32x4;

union FragAB { uint4 u[2]; v16bf v; };

__device__ __forceinline__ unsigned short f2bf(float f) {
  unsigned int u = __float_as_uint(f);
  unsigned int r = u + 0x7fffu + ((u >> 16) & 1u);   // RNE
  return (unsigned short)(r >> 16);
}

// fixed-order block reductions (deterministic)
__device__ __forceinline__ float blk_sum(float v, float* sm, int nw) {
  int lane = threadIdx.x & 31, w = threadIdx.x >> 5;
#pragma unroll
  for (int o = 16; o; o >>= 1) v += __shfl_down(v, o);
  if (lane == 0) sm[w] = v;
  __syncthreads();
  float r = 0.f;
  for (int i = 0; i < nw; ++i) r += sm[i];
  __syncthreads();
  return r;
}
__device__ __forceinline__ float blk_max(float v, float* sm, int nw) {
  int lane = threadIdx.x & 31, w = threadIdx.x >> 5;
#pragma unroll
  for (int o = 16; o; o >>= 1) v = fmaxf(v, __shfl_down(v, o));
  if (lane == 0) sm[w] = v;
  __syncthreads();
  float r = -__builtin_inff();
  for (int i = 0; i < nw; ++i) r = fmaxf(r, sm[i]);
  __syncthreads();
  return r;
}

// ---------------------------------------------------------------------------
// TDM descriptor build + issue: move a 128-row x 64-byte tile of a row-major
// bf16 matrix (row stride K elements) into LDS with 32B padding per 64B row
// (LDS row stride 96B = 48 halves). CDNA5 ISA ch.8:
//  G0: [1:0]=count=1, [63:32]=lds_addr, [120:64]=global_addr, [127:126]=2
//  G1: [17:16]=data_size(3 => 8B), [20]=pad_enable, [24:22]=pad_interval
//      (3 => every 16 dwords), [31:25]=pad_amount (7 => 8 dwords),
//      [79:48]=tensor_dim0, [111:80]=tensor_dim1, [127:112]=tile_dim0(8),
//      [143:128]=tile_dim1(128), [207:160]=tensor_dim0_stride (8B units)
// This toolchain exposes the 6-arg builtin (extra int32x8 descriptor words;
// zero = unused for a 2D tile).
// ---------------------------------------------------------------------------
__device__ __forceinline__ void tdm_load_tile(const unsigned short* gsrc,
                                              const unsigned short* lds_dst,
                                              int K /*elements per row*/) {
  const unsigned long long ga = (unsigned long long)(size_t)gsrc;
  const unsigned int lds_off = (unsigned int)(size_t)lds_dst;  // LDS aperture: low 32b
  u32x4 g0;
  g0[0] = 1u;                                   // count=1, load, user mode
  g0[1] = lds_off;                              // lds_addr
  g0[2] = (unsigned int)ga;                     // global_addr[31:0]
  g0[3] = (unsigned int)(ga >> 32) | (2u << 30);// global_addr[56:32] | type=2

  const unsigned int stride0 = (unsigned int)K / 4u;   // row stride, 8B units
  const unsigned int td0 = 0x100000u, td1 = 0x100000u; // loose OOB bounds
  i32x8 g1;
  g1[0] = (3 << 16) | (1 << 20) | (3 << 22) | (7 << 25);
  g1[1] = (int)((td0 & 0xFFFFu) << 16);
  g1[2] = (int)((td0 >> 16) | ((td1 & 0xFFFFu) << 16));
  g1[3] = (int)((td1 >> 16) | (8u << 16));      // tile_dim0 = 8 (x8B = 64B)
  g1[4] = 128;                                  // tile_dim1 = 128 rows
  g1[5] = (int)stride0;
  g1[6] = 0;
  g1[7] = 0;

  const i32x4 z4 = {0, 0, 0, 0};                // groups 2/3: dims unused (2D)
  const i32x8 z8 = {0, 0, 0, 0, 0, 0, 0, 0};
  __builtin_amdgcn_tensor_load_to_lds(g0, g1, z4, z4, z8, 0);
}

// ---------------------------------------------------------------------------
// fp32 -> bf16 bulk convert (float4 / ushort4)
// ---------------------------------------------------------------------------
__global__ void k_cvt_bf16(const float* __restrict__ in,
                           unsigned short* __restrict__ out, long long n4) {
  long long i = (long long)blockIdx.x * blockDim.x + threadIdx.x;
  if (i >= n4) return;
  float4 v = ((const float4*)in)[i];
  ushort4 o;
  o.x = f2bf(v.x); o.y = f2bf(v.y); o.z = f2bf(v.z); o.w = f2bf(v.w);
  ((ushort4*)out)[i] = o;
}

// ---------------------------------------------------------------------------
// x[row] = tok_emb[idx[row]] + pos_emb[row % T]   (fp32)
// ---------------------------------------------------------------------------
__global__ __launch_bounds__(128) void k_embed(const int* __restrict__ idx,
                                               const float* __restrict__ te,
                                               const float* __restrict__ pe,
                                               float* __restrict__ x) {
  const int row = blockIdx.x, tid = threadIdx.x;
  const int tok = idx[row], t = row & (Tdim - 1);
  float4 a = ((const float4*)(te + (size_t)tok * Cdim))[tid];
  float4 p = ((const float4*)(pe + (size_t)t   * Cdim))[tid];
  a.x += p.x; a.y += p.y; a.z += p.z; a.w += p.w;
  ((float4*)(x + (size_t)row * Cdim))[tid] = a;
}

// ---------------------------------------------------------------------------
// LayerNorm over C=512, writes bf16 (feeds WMMA A operand directly)
// ---------------------------------------------------------------------------
__global__ __launch_bounds__(128) void k_layernorm(const float* __restrict__ x,
                                                   const float* __restrict__ g,
                                                   const float* __restrict__ b,
                                                   unsigned short* __restrict__ out) {
  const int row = blockIdx.x, tid = threadIdx.x;
  const float4 v = ((const float4*)(x + (size_t)row * Cdim))[tid];
  float s1 = v.x + v.y + v.z + v.w;
  float s2 = v.x * v.x + v.y * v.y + v.z * v.z + v.w * v.w;
  __shared__ float sm1[4], sm2[4];
  const int lane = tid & 31, w = tid >> 5;
#pragma unroll
  for (int o = 16; o; o >>= 1) { s1 += __shfl_down(s1, o); s2 += __shfl_down(s2, o); }
  if (lane == 0) { sm1[w] = s1; sm2[w] = s2; }
  __syncthreads();
  float t1 = sm1[0] + sm1[1] + sm1[2] + sm1[3];
  float t2 = sm2[0] + sm2[1] + sm2[2] + sm2[3];
  const float mu  = t1 * (1.f / Cdim);
  const float var = t2 * (1.f / Cdim) - mu * mu;
  const float inv = rsqrtf(var + 1e-5f);
  const float4 gv = ((const float4*)g)[tid];
  const float4 bv = ((const float4*)b)[tid];
  ushort4 o4;
  o4.x = f2bf((v.x - mu) * inv * gv.x + bv.x);
  o4.y = f2bf((v.y - mu) * inv * gv.y + bv.y);
  o4.z = f2bf((v.z - mu) * inv * gv.z + bv.z);
  o4.w = f2bf((v.w - mu) * inv * gv.w + bv.w);
  ((ushort4*)(out + (size_t)row * Cdim))[tid] = o4;
}

// ---------------------------------------------------------------------------
// bf16 [T,C] -> [C,T] transpose (per batch), for the P@V GEMM B operand
// ---------------------------------------------------------------------------
__global__ void k_transpose(const unsigned short* __restrict__ in,
                            unsigned short* __restrict__ out) {
  __shared__ unsigned short tile[32][33];
  const int z = blockIdx.z;
  in  += (size_t)z * Tdim * Cdim;
  out += (size_t)z * Cdim * Tdim;
  const int r0 = blockIdx.x * 32, c0 = blockIdx.y * 32;
  const int tx = threadIdx.x, ty = threadIdx.y;
#pragma unroll
  for (int j = 0; j < 4; ++j)
    tile[ty + j * 8][tx] = in[(size_t)(r0 + ty + j * 8) * Cdim + c0 + tx];
  __syncthreads();
#pragma unroll
  for (int j = 0; j < 4; ++j)
    out[(size_t)(c0 + ty + j * 8) * Tdim + r0 + tx] = tile[tx][ty + j * 8];
}

// ---------------------------------------------------------------------------
// WMMA bf16 TN GEMM:  D[M,N] = alpha * A[M,K] * W[N,K]^T  (+bias/relu/mask/res)
// Block: 256 threads = 8 waves; block tile 128x128; wave tile 32x64
// (2x4 fragments of v_wmma_f32_16x16x32_bf16). K-step 32.
// Data movement: double-buffered LDS, tiles fetched by the Tensor Data Mover
// (wave 0 issues tensor_load_to_lds, TENSORcnt + workgroup barrier publish),
// so copy overlaps compute with no VGPR round trip.
// Per-lane fragment layout per CDNA5 ISA §7.12.2:
//   A/B: row = lane&15; halves = K in chunks [half*8..+7] and [16+half*8..+7]
//   C/D: m = r + 8*(lane>=16), n = lane&15
// ---------------------------------------------------------------------------
#define GEMM_RELU 1
#define GEMM_MASK 2
#define GEMM_RES  4

template <bool OUT_BF16>
__global__ __launch_bounds__(256)
void k_gemm(const unsigned short* __restrict__ A,
            const unsigned short* __restrict__ W,
            void* __restrict__ Dv,
            const float* __restrict__ bias,
            int M, int N, int K,
            long long sA, long long sW, long long sD,
            float alpha, int flags) {
  __shared__ alignas(16) unsigned short As[2][128 * 48];
  __shared__ alignas(16) unsigned short Bs[2][128 * 48];
  const int bz = blockIdx.z;
  A += (size_t)bz * (size_t)sA;
  W += (size_t)bz * (size_t)sW;
  const int m0 = blockIdx.y * 128, n0 = blockIdx.x * 128;
  const int tid = threadIdx.x, lane = tid & 31, wave = tid >> 5;
  const int wm = wave & 3, wn = wave >> 2;   // 4 waves along M, 2 along N
  const int lm = lane & 15, lh = lane >> 4;

  v8f acc[2][4];
#pragma unroll
  for (int i = 0; i < 2; ++i)
#pragma unroll
    for (int j = 0; j < 4; ++j)
#pragma unroll
      for (int e = 0; e < 8; ++e) acc[i][j][e] = 0.f;

  const int nk = K >> 5;                      // K/32 steps
  if (wave == 0) {                            // prologue: fill buffer 0
    tdm_load_tile(&A[(size_t)m0 * K], &As[0][0], K);
    tdm_load_tile(&W[(size_t)n0 * K], &Bs[0][0], K);
  }

  for (int kt = 0; kt < nk; ++kt) {
    const int cur = kt & 1;
    if (wave == 0) __builtin_amdgcn_s_wait_tensorcnt(0);
    __syncthreads();                          // cur filled; prev buffer free
    if (wave == 0 && (kt + 1) < nk) {         // prefetch next tile via TDM
      const int kn = (kt + 1) << 5;
      tdm_load_tile(&A[(size_t)m0 * K + kn], &As[cur ^ 1][0], K);
      tdm_load_tile(&W[(size_t)n0 * K + kn], &Bs[cur ^ 1][0], K);
    }

    FragAB fa[2], fb[4];
#pragma unroll
    for (int i = 0; i < 2; ++i) {
      const unsigned short* base = &As[cur][(wm * 32 + i * 16 + lm) * 48 + lh * 8];
      fa[i].u[0] = *(const uint4*)base;
      fa[i].u[1] = *(const uint4*)(base + 16);
    }
#pragma unroll
    for (int j = 0; j < 4; ++j) {
      const unsigned short* base = &Bs[cur][(wn * 64 + j * 16 + lm) * 48 + lh * 8];
      fb[j].u[0] = *(const uint4*)base;
      fb[j].u[1] = *(const uint4*)(base + 16);
    }
#pragma unroll
    for (int i = 0; i < 2; ++i)
#pragma unroll
      for (int j = 0; j < 4; ++j)
        acc[i][j] = __builtin_amdgcn_wmma_f32_16x16x32_bf16(
            false, fa[i].v, false, fb[j].v, (short)0, acc[i][j], false, false);
  }

#pragma unroll
  for (int i = 0; i < 2; ++i)
#pragma unroll
    for (int j = 0; j < 4; ++j)
#pragma unroll
      for (int r = 0; r < 8; ++r) {
        const int gm = m0 + wm * 32 + i * 16 + lh * 8 + r;
        const int gn = n0 + wn * 64 + j * 16 + lm;
        const size_t off = (size_t)bz * (size_t)sD + (size_t)gm * N + gn;
        float val = acc[i][j][r] * alpha;
        if (bias) val += bias[gn];
        if (flags & GEMM_RELU) val = fmaxf(val, 0.f);
        if ((flags & GEMM_MASK) && (gn > gm)) val = -__builtin_inff();
        if constexpr (OUT_BF16) {
          ((unsigned short*)Dv)[off] = f2bf(val);
        } else {
          float o = val;
          if (flags & GEMM_RES) o += ((const float*)Dv)[off];
          ((float*)Dv)[off] = o;
        }
      }
}

// ---------------------------------------------------------------------------
// Causal softmax over a 2048-wide score row; writes bf16 probabilities.
// Masked entries arrive as -inf -> exp == 0.
// ---------------------------------------------------------------------------
__global__ __launch_bounds__(256) void k_softmax(const float* __restrict__ wei,
                                                 unsigned short* __restrict__ P) {
  __shared__ float sm[8];
  const int row = blockIdx.x, tid = threadIdx.x;
  const float* r = wei + (size_t)row * Tdim;
  float v[8];
  float mx = -__builtin_inff();
#pragma unroll
  for (int j = 0; j < 8; ++j) { v[j] = r[tid + j * 256]; mx = fmaxf(mx, v[j]); }
  mx = blk_max(mx, sm, 8);
  float s = 0.f;
#pragma unroll
  for (int j = 0; j < 8; ++j) { v[j] = expf(v[j] - mx); s += v[j]; }
  s = blk_sum(s, sm, 8);
  const float inv = 1.f / s;
  unsigned short* pr = P + (size_t)row * Tdim;
#pragma unroll
  for (int j = 0; j < 8; ++j) pr[tid + j * 256] = f2bf(v[j] * inv);
}

// ---------------------------------------------------------------------------
// Cross-entropy: per-row loss, then deterministic reduce to scalar
// ---------------------------------------------------------------------------
__global__ __launch_bounds__(256) void k_ce(const float* __restrict__ logits,
                                            const int* __restrict__ tgt,
                                            float* __restrict__ rowloss) {
  __shared__ float sm[8];
  const int row = blockIdx.x, tid = threadIdx.x;
  const float* r = logits + (size_t)row * Vdim;
  float mx = -__builtin_inff();
  for (int i = tid; i < Vdim; i += 256) mx = fmaxf(mx, r[i]);
  mx = blk_max(mx, sm, 8);
  float s = 0.f;
  for (int i = tid; i < Vdim; i += 256) s += expf(r[i] - mx);
  s = blk_sum(s, sm, 8);
  if (tid == 0) rowloss[row] = mx + logf(s) - r[tgt[row]];
}

__global__ __launch_bounds__(256) void k_loss(const float* __restrict__ rowloss,
                                              float* __restrict__ loss) {
  __shared__ float sm[8];
  float s = 0.f;
  for (int i = threadIdx.x; i < Mrows; i += 256) s += rowloss[i];
  s = blk_sum(s, sm, 8);
  if (threadIdx.x == 0) *loss = s * (1.f / Mrows);
}

// ---------------------------------------------------------------------------
// Host orchestration
// ---------------------------------------------------------------------------
extern "C" void kernel_launch(void* const* d_in, const int* in_sizes, int n_in,
                              void* d_out, int out_size, void* d_ws, size_t ws_size,
                              hipStream_t stream) {
  (void)in_sizes; (void)n_in; (void)out_size; (void)ws_size;
  const int*   idx   = (const int*)d_in[0];
  const int*   tgt   = (const int*)d_in[1];
  const float* te    = (const float*)d_in[2];
  const float* pe    = (const float*)d_in[3];
  const float* Wk    = (const float*)d_in[4];
  const float* Wq    = (const float*)d_in[5];
  const float* Wv    = (const float*)d_in[6];
  const float* W1    = (const float*)d_in[7];
  const float* b1    = (const float*)d_in[8];
  const float* W2    = (const float*)d_in[9];
  const float* b2    = (const float*)d_in[10];
  const float* g1    = (const float*)d_in[11];
  const float* be1   = (const float*)d_in[12];
  const float* g2    = (const float*)d_in[13];
  const float* be2   = (const float*)d_in[14];
  const float* Wlm   = (const float*)d_in[15];
  const float* blm   = (const float*)d_in[16];

  // workspace carve (256B aligned)
  char* p = (char*)d_ws;
  auto carve = [&](size_t bytes) -> void* {
    void* r = (void*)p; p += (bytes + 255) & ~(size_t)255; return r;
  };
  float*          x    = (float*)         carve((size_t)Mrows * Cdim * 4);
  unsigned short* h    = (unsigned short*)carve((size_t)Mrows * Cdim * 2);
  unsigned short* qb   = (unsigned short*)carve((size_t)Mrows * Cdim * 2);
  unsigned short* kb   = (unsigned short*)carve((size_t)Mrows * Cdim * 2);
  unsigned short* vb   = (unsigned short*)carve((size_t)Mrows * Cdim * 2);
  unsigned short* vT   = (unsigned short*)carve((size_t)Mrows * Cdim * 2);
  float*          wei  = (float*)         carve((size_t)Bdim * Tdim * Tdim * 4);
  unsigned short* P    = (unsigned short*)carve((size_t)Bdim * Tdim * Tdim * 2);
  unsigned short* h2   = (unsigned short*)carve((size_t)Mrows * Cdim * 2);
  unsigned short* ff   = (unsigned short*)carve((size_t)Mrows * Fdim * 2);
  unsigned short* xb   = (unsigned short*)carve((size_t)Mrows * Cdim * 2);
  unsigned short* Wqb  = (unsigned short*)carve((size_t)Cdim * Cdim * 2);
  unsigned short* Wkb  = (unsigned short*)carve((size_t)Cdim * Cdim * 2);
  unsigned short* Wvb  = (unsigned short*)carve((size_t)Cdim * Cdim * 2);
  unsigned short* W1b  = (unsigned short*)carve((size_t)Fdim * Cdim * 2);
  unsigned short* W2b  = (unsigned short*)carve((size_t)Cdim * Fdim * 2);
  unsigned short* Wlmb = (unsigned short*)carve((size_t)Vdim * Cdim * 2);
  float*          rls  = (float*)         carve((size_t)Mrows * 4);

  auto cvt = [&](const float* in, unsigned short* out, long long n) {
    long long n4 = n / 4;
    k_cvt_bf16<<<(unsigned)((n4 + 255) / 256), 256, 0, stream>>>(in, out, n4);
  };
  // weights -> bf16
  cvt(Wq,  Wqb,  (long long)Cdim * Cdim);
  cvt(Wk,  Wkb,  (long long)Cdim * Cdim);
  cvt(Wv,  Wvb,  (long long)Cdim * Cdim);
  cvt(W1,  W1b,  (long long)Fdim * Cdim);
  cvt(W2,  W2b,  (long long)Cdim * Fdim);
  cvt(Wlm, Wlmb, (long long)Vdim * Cdim);

  // embeddings, LN1
  k_embed<<<Mrows, 128, 0, stream>>>(idx, te, pe, x);
  k_layernorm<<<Mrows, 128, 0, stream>>>(x, g1, be1, h);

  // q,k,v projections: [8192,512] = h * W^T
  const dim3 gQKV(Cdim / 128, Mrows / 128, 1);
  k_gemm<true><<<gQKV, 256, 0, stream>>>(h, Wqb, qb, nullptr,
      Mrows, Cdim, Cdim, 0, 0, 0, 1.f, 0);
  k_gemm<true><<<gQKV, 256, 0, stream>>>(h, Wkb, kb, nullptr,
      Mrows, Cdim, Cdim, 0, 0, 0, 1.f, 0);
  k_gemm<true><<<gQKV, 256, 0, stream>>>(h, Wvb, vb, nullptr,
      Mrows, Cdim, Cdim, 0, 0, 0, 1.f, 0);

  // v -> vT per batch (for P@V as TN gemm)
  k_transpose<<<dim3(Tdim / 32, Cdim / 32, Bdim), dim3(32, 8), 0, stream>>>(vb, vT);

  // S = (q k^T) * C^-0.5, causal masked   [per batch 2048x2048, K=512]
  const float scale = 0.044194173824159216f; // 512^-0.5
  k_gemm<false><<<dim3(Tdim / 128, Tdim / 128, Bdim), 256, 0, stream>>>(
      qb, kb, wei, nullptr, Tdim, Tdim, Cdim,
      (long long)Tdim * Cdim, (long long)Tdim * Cdim, (long long)Tdim * Tdim,
      scale, GEMM_MASK);

  // softmax rows -> P (bf16)
  k_softmax<<<Bdim * Tdim, 256, 0, stream>>>(wei, P);

  // x += P @ V      [per batch 2048x512, K=2048]
  k_gemm<false><<<dim3(Cdim / 128, Tdim / 128, Bdim), 256, 0, stream>>>(
      P, vT, x, nullptr, Tdim, Cdim, Tdim,
      (long long)Tdim * Tdim, (long long)Cdim * Tdim, (long long)Tdim * Cdim,
      1.f, GEMM_RES);

  // LN2, FFN
  k_layernorm<<<Mrows, 128, 0, stream>>>(x, g2, be2, h2);
  k_gemm<true><<<dim3(Fdim / 128, Mrows / 128, 1), 256, 0, stream>>>(
      h2, W1b, ff, b1, Mrows, Fdim, Cdim, 0, 0, 0, 1.f, GEMM_RELU);
  k_gemm<false><<<dim3(Cdim / 128, Mrows / 128, 1), 256, 0, stream>>>(
      ff, W2b, x, b2, Mrows, Cdim, Fdim, 0, 0, 0, 1.f, GEMM_RES);

  // LM head: logits (fp32) straight into d_out
  cvt(x, xb, (long long)Mrows * Cdim);
  k_gemm<false><<<dim3(Vdim / 128, Mrows / 128, 1), 256, 0, stream>>>(
      xb, Wlmb, d_out, blm, Mrows, Vdim, Cdim, 0, 0, 0, 1.f, 0);

  // cross-entropy
  k_ce<<<Mrows, 256, 0, stream>>>((const float*)d_out, tgt, rls);
  k_loss<<<1, 256, 0, stream>>>(rls, (float*)d_out + (size_t)Mrows * Vdim);
}
